// GraphNet_80547816669403
// MI455X (gfx1250) — matrix operand order repaired and checked by reading, not compiled, not run on hardware
//
#include <hip/hip_runtime.h>
#include <hip/hip_bf16.h>

#define BATCH 128
#define NNODE 188
#define PFEAT 16
#define NRE   17578            // NNODE*(NNODE-1)/2
#define HID   128
#define H2    64
#define DE    5
#define DO    6
#define NT    2

typedef _Float16 f16;
typedef __attribute__((ext_vector_type(16))) _Float16 v16h;
typedef __attribute__((ext_vector_type(8)))  _Float16 v8h;
typedef __attribute__((ext_vector_type(8)))  float    v8f;
typedef __attribute__((ext_vector_type(4)))  float    v4f;

// gfx1250 has a hardware V_TANH_F32 (TRANS op). Prefer the builtin; otherwise
// use a branch-free rational form on the native exp (v_exp_f32).
__device__ __forceinline__ float fast_tanh(float x) {
#if __has_builtin(__builtin_amdgcn_tanhf)
  return __builtin_amdgcn_tanhf(x);
#else
  float e = __expf(x + x);            // v_mul + v_exp (native, no branches)
  return 1.0f - 2.0f / (e + 1.0f);    // saturates correctly at +/-1 for |x| large
#endif
}

__device__ __forceinline__ v8f wmma_f16(v16h a, v16h b, v8f c) {
  // D = A(16x32 f16) x B(32x16 f16) + C(16x16 f32)
  return __builtin_amdgcn_wmma_f32_16x16x32_f16(false, a, false, b, (short)0, c, false, false);
}

// B fragment (32x16, f16): lane: N = lane%16, element e: K = kbase + (e>=8?16:0) + (lane/16)*8 + (e&7)
// w laid out [n*ldk + k], padded so all reads are in-range / zero.
__device__ __forceinline__ v16h load_bfrag(const f16* w, int n, int ldk, int kbase, int half) {
  const f16* p = w + n * ldk + kbase + half * 8;
  v8h lo = *(const v8h*)p;
  v8h hi = *(const v8h*)(p + 16);
  return __builtin_shufflevector(lo, hi, 0,1,2,3,4,5,6,7,8,9,10,11,12,13,14,15);
}

// A fragment (16x32, f16) from column-major LDS tile h[(col k)*16 + row m]:
// element e: K = kbase + (e>=8?16:0) + half*8 + (e&7), row M = lane%16
__device__ __forceinline__ v16h load_afrag(const f16* h, int m, int kbase, int half) {
  v16h r;
#pragma unroll
  for (int e = 0; e < 16; ++e) {
    int k = kbase + ((e >> 3) << 4) + half * 8 + (e & 7);
    r[e] = h[k * 16 + m];
  }
  return r;
}

// Layer1: h1(16x128) = tanh(A1(16x32) @ W1^T + b1) written col-major to LDS tile h.
// lw1 is f16[128*32] (K zero-padded), lb1 f32[128].
__device__ __forceinline__ void mlp_layer1(v16h a1, f16* h, const f16* lw1, const float* lb1, int lane) {
  const int m = lane & 15, half = lane >> 4;
#pragma unroll
  for (int t = 0; t < 8; ++t) {
    const int n = t * 16 + m;
    const float bb = lb1[n];
    v8f c = {bb, bb, bb, bb, bb, bb, bb, bb};
    v16h bf = load_bfrag(lw1, n, 32, 0, half);
    c = wmma_f16(a1, bf, c);
    v8h pk;
#pragma unroll
    for (int v = 0; v < 8; ++v) pk[v] = (f16)fast_tanh(c[v]);
    *(v8h*)(h + n * 16 + half * 8) = pk;   // col n, rows half*8 .. half*8+7
  }
}

// Layers 2+3: h2(16x64)=tanh(h1 @ W2^T + b2); returns c3 = h2 @ W3^T + b3 (pre-activation).
// lw2 f16[64*128], lw3 f16[16*64] (rows >= out_dim zeroed), lb3 f32[16] (padded).
__device__ __forceinline__ v8f mlp_tail(f16* h, const f16* lw2, const float* lb2,
                                        const f16* lw3, const float* lb3, int lane) {
  const int m = lane & 15, half = lane >> 4;
  v8f c2[4];
#pragma unroll
  for (int t = 0; t < 4; ++t) {
    const float bb = lb2[t * 16 + m];
    c2[t] = (v8f){bb, bb, bb, bb, bb, bb, bb, bb};
  }
#pragma unroll
  for (int kc = 0; kc < 4; ++kc) {
    v16h a = load_afrag(h, m, 32 * kc, half);
#pragma unroll
    for (int t = 0; t < 4; ++t) {
      v16h bf = load_bfrag(lw2, t * 16 + m, 128, 32 * kc, half);
      c2[t] = wmma_f16(a, bf, c2[t]);
    }
  }
#pragma unroll
  for (int t = 0; t < 4; ++t) {
    const int n = t * 16 + m;
    v8h pk;
#pragma unroll
    for (int v = 0; v < 8; ++v) pk[v] = (f16)fast_tanh(c2[t][v]);
    *(v8h*)(h + n * 16 + half * 8) = pk;
  }
  const float b3 = lb3[m];
  v8f c3 = {b3, b3, b3, b3, b3, b3, b3, b3};
#pragma unroll
  for (int kc = 0; kc < 2; ++kc) {
    v16h a = load_afrag(h, m, 32 * kc, half);
    v16h bf = load_bfrag(lw3, m, 64, 32 * kc, half);
    c3 = wmma_f16(a, bf, c3);
  }
  return c3;
}

// ---------------- setup: pair decode + zero accumulators ----------------
__global__ void k_setup(int* pi, int* pj, float* Ebar, float* Osum) {
  const int idx = blockIdx.x * blockDim.x + threadIdx.x;
  if (idx < NRE) {
    const float tn = 2.0f * NNODE - 1.0f;
    float disc = tn * tn - 8.0f * (float)idx;
    int i = (int)floorf((tn - sqrtf(disc)) * 0.5f);
    if (i < 0) i = 0;
    if (i > NNODE - 2) i = NNODE - 2;
    // off(i) = i*(NNODE-1) - i*(i-1)/2
    #define OFF(a) ((a) * (NNODE - 1) - ((a) * ((a) - 1)) / 2)
    while (i > 0 && OFF(i) > idx) --i;
    while (i < NNODE - 2 && OFF(i + 1) <= idx) ++i;
    pi[idx] = i;
    pj[idx] = i + 1 + (idx - OFF(i));
    #undef OFF
  }
  if (idx < BATCH * NNODE * DE) Ebar[idx] = 0.0f;
  if (idx < BATCH * DO)         Osum[idx] = 0.0f;
}

// ---------------- edge MLP + scatter ----------------
__global__ __launch_bounds__(256) void k_edge(const float* __restrict__ x,
                                              const int* __restrict__ pi, const int* __restrict__ pj,
                                              const float* w1, const float* b1,
                                              const float* w2, const float* b2,
                                              const float* w3, const float* b3,
                                              float* __restrict__ Ebar, int numTiles) {
  __shared__ __align__(16) f16 lw1[128 * 32];
  __shared__ __align__(16) f16 lw2[64 * 128];
  __shared__ __align__(16) f16 lw3[16 * 64];
  __shared__ float lb1[128], lb2[64], lb3[16];
  __shared__ __align__(16) f16 hbuf[8][128 * 16];

  const int tid = threadIdx.x;
  for (int idx = tid; idx < 128 * 32; idx += 256) {
    int n = idx >> 5, k = idx & 31;
    lw1[idx] = (k < 16) ? (f16)w1[n * 16 + k] : (f16)0.0f;
  }
  for (int idx = tid; idx < 64 * 128; idx += 256) lw2[idx] = (f16)w2[idx];
  for (int idx = tid; idx < 16 * 64; idx += 256) lw3[idx] = (idx < DE * 64) ? (f16)w3[idx] : (f16)0.0f;
  for (int idx = tid; idx < 128; idx += 256) lb1[idx] = b1[idx];
  for (int idx = tid; idx < 64; idx += 256)  lb2[idx] = b2[idx];
  for (int idx = tid; idx < 16; idx += 256)  lb3[idx] = (idx < DE) ? b3[idx] : 0.0f;
  __syncthreads();

  const int lane = tid & 31;
  const int m = lane & 15, half = lane >> 4;
  f16* h = hbuf[tid >> 5];
  const int gwave = blockIdx.x * 8 + (tid >> 5);
  const int nWaves = gridDim.x * 8;

  for (int tile = gwave; tile < numTiles; tile += nWaves) {
    const int row = tile * 16 + m;          // row in [0, BATCH*NRE)
    const int b = row / NRE;
    const int r = row - b * NRE;
    const int ii = pi[r], jj = pj[r];
    const float* xi = x + ((b * NNODE + ii) * PFEAT + half * 8);
    const float* xj = x + ((b * NNODE + jj) * PFEAT + half * 8);
    v4f xi0 = *(const v4f*)xi, xi1 = *(const v4f*)(xi + 4);
    v4f xj0 = *(const v4f*)xj, xj1 = *(const v4f*)(xj + 4);
    v16h a1;
#pragma unroll
    for (int e = 0; e < 4; ++e) {
      a1[e]      = (f16)(xi0[e] - xj0[e]);
      a1[4 + e]  = (f16)(xi1[e] - xj1[e]);
      a1[8 + e]  = (f16)0.0f;
      a1[12 + e] = (f16)0.0f;
    }
    mlp_layer1(a1, h, lw1, lb1, lane);
    v8f c3 = mlp_tail(h, lw2, lb2, lw3, lb3, lane);
    if (m < DE) {
#pragma unroll
      for (int v = 0; v < 8; ++v) {
        const int rrow = tile * 16 + v + half * 8;
        const int bb = rrow / NRE;
        const int rr = rrow - bb * NRE;
        const int recv = pi[rr];
        atomicAdd(Ebar + ((bb * NNODE + recv) * DE + m), fast_tanh(c3[v]));
      }
    }
  }
}

// ---------------- node MLP + per-batch sum ----------------
__global__ __launch_bounds__(256) void k_node(const float* __restrict__ x,
                                              const float* __restrict__ Ebar,
                                              const float* w1, const float* b1,
                                              const float* w2, const float* b2,
                                              const float* w3, const float* b3,
                                              float* __restrict__ Osum, int numTiles) {
  __shared__ __align__(16) f16 lw1[128 * 32];
  __shared__ __align__(16) f16 lw2[64 * 128];
  __shared__ __align__(16) f16 lw3[16 * 64];
  __shared__ float lb1[128], lb2[64], lb3[16];
  __shared__ __align__(16) f16 hbuf[8][128 * 16];

  const int tid = threadIdx.x;
  const int KIN = PFEAT + DE; // 21
  for (int idx = tid; idx < 128 * 32; idx += 256) {
    int n = idx >> 5, k = idx & 31;
    lw1[idx] = (k < KIN) ? (f16)w1[n * KIN + k] : (f16)0.0f;
  }
  for (int idx = tid; idx < 64 * 128; idx += 256) lw2[idx] = (f16)w2[idx];
  for (int idx = tid; idx < 16 * 64; idx += 256) lw3[idx] = (idx < DO * 64) ? (f16)w3[idx] : (f16)0.0f;
  for (int idx = tid; idx < 128; idx += 256) lb1[idx] = b1[idx];
  for (int idx = tid; idx < 64; idx += 256)  lb2[idx] = b2[idx];
  for (int idx = tid; idx < 16; idx += 256)  lb3[idx] = (idx < DO) ? b3[idx] : 0.0f;
  __syncthreads();

  const int lane = tid & 31;
  const int m = lane & 15, half = lane >> 4;
  f16* h = hbuf[tid >> 5];
  const int gwave = blockIdx.x * 8 + (tid >> 5);
  const int nWaves = gridDim.x * 8;

  for (int tile = gwave; tile < numTiles; tile += nWaves) {
    const int row = tile * 16 + m;          // row in [0, BATCH*NNODE)
    const int b = row / NNODE;
    const int node = row - b * NNODE;
    const float* xr = x + (b * NNODE + node) * PFEAT + half * 8;
    const float* er = Ebar + (b * NNODE + node) * DE;
    v4f x0 = *(const v4f*)xr, x1 = *(const v4f*)(xr + 4);
    v16h a1;
#pragma unroll
    for (int e = 0; e < 4; ++e) {
      a1[e]     = (f16)x0[e];
      a1[4 + e] = (f16)x1[e];
    }
#pragma unroll
    for (int e = 8; e < 16; ++e) {
      // K = 16 + half*8 + (e-8); real only when half==0 and (e-8) < DE
      int kk = e - 8;
      a1[e] = (half == 0 && kk < DE) ? (f16)er[kk] : (f16)0.0f;
    }
    mlp_layer1(a1, h, lw1, lb1, lane);
    v8f c3 = mlp_tail(h, lw2, lb2, lw3, lb3, lane);
    if (m < DO) {
#pragma unroll
      for (int v = 0; v < 8; ++v) {
        const int rrow = tile * 16 + v + half * 8;
        const int bb = rrow / NNODE;
        atomicAdd(Osum + bb * DO + m, fast_tanh(c3[v]));
      }
    }
  }
}

// ---------------- readout MLP (final linear, no tanh) ----------------
__global__ __launch_bounds__(256) void k_readout(const float* __restrict__ Osum,
                                                 const float* w1, const float* b1,
                                                 const float* w2, const float* b2,
                                                 const float* w3, const float* b3,
                                                 float* __restrict__ out) {
  __shared__ __align__(16) f16 lw1[128 * 32];
  __shared__ __align__(16) f16 lw2[64 * 128];
  __shared__ __align__(16) f16 lw3[16 * 64];
  __shared__ float lb1[128], lb2[64], lb3[16];
  __shared__ __align__(16) f16 hbuf[8][128 * 16];

  const int tid = threadIdx.x;
  for (int idx = tid; idx < 128 * 32; idx += 256) {
    int n = idx >> 5, k = idx & 31;
    lw1[idx] = (k < DO) ? (f16)w1[n * DO + k] : (f16)0.0f;
  }
  for (int idx = tid; idx < 64 * 128; idx += 256) lw2[idx] = (f16)w2[idx];
  for (int idx = tid; idx < 16 * 64; idx += 256) lw3[idx] = (idx < NT * 64) ? (f16)w3[idx] : (f16)0.0f;
  for (int idx = tid; idx < 128; idx += 256) lb1[idx] = b1[idx];
  for (int idx = tid; idx < 64; idx += 256)  lb2[idx] = b2[idx];
  for (int idx = tid; idx < 16; idx += 256)  lb3[idx] = (idx < NT) ? b3[idx] : 0.0f;
  __syncthreads();

  const int lane = tid & 31;
  const int m = lane & 15, half = lane >> 4;
  f16* h = hbuf[tid >> 5];
  const int tile = tid >> 5;               // 8 tiles of 16 rows = 128 batch rows

  const int row = tile * 16 + m;
  v16h a1;
#pragma unroll
  for (int e = 0; e < 8; ++e) {
    int k = half * 8 + e;
    a1[e] = (k < DO) ? (f16)Osum[row * DO + k] : (f16)0.0f;
    a1[8 + e] = (f16)0.0f;
  }
  mlp_layer1(a1, h, lw1, lb1, lane);
  v8f c3 = mlp_tail(h, lw2, lb2, lw3, lb3, lane);   // final linear incl. bias
  if (m < NT) {
#pragma unroll
    for (int v = 0; v < 8; ++v) {
      const int rrow = tile * 16 + v + half * 8;
      out[rrow * NT + m] = c3[v];
    }
  }
}

extern "C" void kernel_launch(void* const* d_in, const int* in_sizes, int n_in,
                              void* d_out, int out_size, void* d_ws, size_t ws_size,
                              hipStream_t stream) {
  const float* x     = (const float*)d_in[0];
  // d_in[1] (RR), d_in[2] (Rr) unused: pair structure is synthesized analytically.
  const float* fr1_w = (const float*)d_in[3];
  const float* fr1_b = (const float*)d_in[4];
  const float* fr2_w = (const float*)d_in[5];
  const float* fr2_b = (const float*)d_in[6];
  const float* fr3_w = (const float*)d_in[7];
  const float* fr3_b = (const float*)d_in[8];
  const float* fo1_w = (const float*)d_in[9];
  const float* fo1_b = (const float*)d_in[10];
  const float* fo2_w = (const float*)d_in[11];
  const float* fo2_b = (const float*)d_in[12];
  const float* fo3_w = (const float*)d_in[13];
  const float* fo3_b = (const float*)d_in[14];
  const float* fc1_w = (const float*)d_in[15];
  const float* fc1_b = (const float*)d_in[16];
  const float* fc2_w = (const float*)d_in[17];
  const float* fc2_b = (const float*)d_in[18];
  const float* fc3_w = (const float*)d_in[19];
  const float* fc3_b = (const float*)d_in[20];

  char* ws = (char*)d_ws;
  size_t off = 0;
  int*   pi   = (int*)(ws + off);   off += ((size_t)NRE * 4 + 255) & ~(size_t)255;
  int*   pj   = (int*)(ws + off);   off += ((size_t)NRE * 4 + 255) & ~(size_t)255;
  float* Ebar = (float*)(ws + off); off += ((size_t)BATCH * NNODE * DE * 4 + 255) & ~(size_t)255;
  float* Osum = (float*)(ws + off); off += ((size_t)BATCH * DO * 4 + 255) & ~(size_t)255;

  const int setupThreads = BATCH * NNODE * DE; // 120320 > NRE
  k_setup<<<(setupThreads + 255) / 256, 256, 0, stream>>>(pi, pj, Ebar, Osum);

  const int edgeTiles = (BATCH * NRE) / 16;    // 140624, exact
  k_edge<<<512, 256, 0, stream>>>(x, pi, pj, fr1_w, fr1_b, fr2_w, fr2_b, fr3_w, fr3_b,
                                  Ebar, edgeTiles);

  const int nodeTiles = (BATCH * NNODE) / 16;  // 1504, exact
  k_node<<<188, 256, 0, stream>>>(x, Ebar, fo1_w, fo1_b, fo2_w, fo2_b, fo3_w, fo3_b,
                                  Osum, nodeTiles);

  k_readout<<<1, 256, 0, stream>>>(Osum, fc1_w, fc1_b, fc2_w, fc2_b, fc3_w, fc3_b,
                                   (float*)d_out);
}
